// Decoder_91293824844343
// MI455X (gfx1250) — matrix operand order repaired and checked by reading, not compiled, not run on hardware
//
#include <hip/hip_runtime.h>
#include <hip/hip_bf16.h>

// ---------------------------------------------------------------------------
// CDNA5 (gfx1250) attention-LSTM decoder.
// GEMMs use v_wmma_f32_16x16x32_bf16. The A (activation) panel is staged in
// LDS once per block (shared by all 8 waves, bank-conflict-free padded rows);
// the B (weight) stream comes from global/L2 with WGP-scope prefetch.
// ---------------------------------------------------------------------------

typedef __bf16 bf16_t;
typedef __attribute__((ext_vector_type(16))) __bf16 v16bf;
typedef __attribute__((ext_vector_type(8)))  __bf16 v8bf;
typedef __attribute__((ext_vector_type(8)))  float  v8f;

#define NB     256   // batch
#define TENC   512   // encoder timesteps
#define KSZ    256   // key size
#define VSZ    256   // value size
#define HSZ    512   // hidden size
#define VOC    35
#define STEPS  300
#define TEXTL  301
#define K1     1280  // [x(512) | ctx(256) | h1(512)]
#define N1     2048  // 4*H
#define K2     768   // [h1(512) | h2(256)]
#define N2     1024  // 4*KS

__device__ __forceinline__ float bf2f(bf16_t x) {
  unsigned short u = __builtin_bit_cast(unsigned short, x);
  unsigned v = ((unsigned)u) << 16;
  return __builtin_bit_cast(float, v);
}
__device__ __forceinline__ bf16_t f2bf(float f) {
  unsigned u = __builtin_bit_cast(unsigned, f);
  unsigned short h = (unsigned short)((u + 0x7FFFu + ((u >> 16) & 1u)) >> 16);
  return __builtin_bit_cast(bf16_t, h);
}
__device__ __forceinline__ float sigf(float x) { return 1.0f / (1.0f + __expf(-x)); }

// --- WMMA fragment loads (wave32; contiguous k order chosen here, same for A and B) ----
// A 16x32 bf16: lane<16 -> row m0+lane, K {0..15}; lane>=16 -> row
// m0+lane-16, K {16..31}. Two contiguous 16B loads per lane.
// (Works identically from LDS: two ds_load_b128 per lane.)
__device__ __forceinline__ v16bf frag_a_from(const bf16_t* p16) {
  union { v16bf v; v8bf h[2]; } u;
  u.h[0] = *reinterpret_cast<const v8bf*>(p16);
  u.h[1] = *reinterpret_cast<const v8bf*>(p16 + 8);
  return u.v;
}
// B 32x16 bf16 with B[k,n] = W[n,k] (row-major W, stride ldb): lane<16 ->
// col n0+lane, K k0..k0+15; lane>=16 -> col n0+lane-16, K k0+16..k0+31.
// 16 consecutive k per lane = 32 contiguous bytes.
__device__ __forceinline__ v16bf load_frag_b(const bf16_t* __restrict__ W,
                                             int ldb, int n0, int k0, int lane) {
  const int half = lane >> 4;
  const int n    = n0 + (lane & 15);
  const bf16_t* p = W + (size_t)n * ldb + k0 + half * 16;
  union { v16bf v; v8bf h[2]; } u;
  u.h[0] = *reinterpret_cast<const v8bf*>(p);
  u.h[1] = *reinterpret_cast<const v8bf*>(p + 8);
  return u.v;
}

__device__ __forceinline__ v8f wmma_bf16(v16bf a, v16bf b, v8f c) {
  return __builtin_amdgcn_wmma_f32_16x16x32_bf16(false, a, false, b,
                                                 (short)0, c, false, false);
}

// C = A(MxK, bf16, row-major, lda==K) * W(NxK, bf16, row-major, ldb==K)^T.
// Block = 256 threads = 8 waves, block tile 32(M) x 256(N); the 32xK A panel
// is staged in LDS once (80KB for K=1280 -- CDNA5 allows 320KB/workgroup).
// Row pad of 8 bf16 makes the per-lane ds_load_b128 fragment reads hit all
// 64 LDS banks exactly twice (conflict-free).
template <int K>
__global__ void wmma_gemm_lds(const bf16_t* __restrict__ A,
                              const bf16_t* __restrict__ B,
                              float* __restrict__ C, int N) {
  constexpr int LDK = K + 8;
  __shared__ bf16_t As[32 * LDK];
  const int tid  = threadIdx.x;
  const int lane = tid & 31;
  const int wv   = tid >> 5;                 // 0..7
  const int nblk = N >> 8;                   // N / 256
  const int m0   = (blockIdx.x / nblk) << 5; // A panel row base
  const int n0   = ((blockIdx.x % nblk) << 8) + (wv << 5);

  // Cooperative stage of the 32xK A panel into padded LDS rows.
  {
    constexpr int CH = (32 * K) / 8;         // 16-byte chunks
    for (int c = tid; c < CH; c += 256) {
      const int row = c / (K / 8);
      const int k8  = (c - row * (K / 8)) * 8;
      *reinterpret_cast<v8bf*>(&As[row * LDK + k8]) =
          *reinterpret_cast<const v8bf*>(A + (size_t)(m0 + row) * K + k8);
    }
  }
  __syncthreads();

  const int half = lane >> 4;
  const int l16  = lane & 15;
  v8f acc00 = {}; v8f acc01 = {}; v8f acc10 = {}; v8f acc11 = {};

  for (int k0 = 0; k0 < K; k0 += 32) {
    if (k0 + 32 < K) {  // WGP-scope prefetch of the next weight K-chunk
      __builtin_prefetch(B + (size_t)(n0 + l16) * K + k0 + 32, 0, 3);
      __builtin_prefetch(B + (size_t)(n0 + 16 + l16) * K + k0 + 32, 0, 3);
    }
    const bf16_t* ap0 = &As[(l16)      * LDK + k0 + half * 16];
    const bf16_t* ap1 = &As[(16 + l16) * LDK + k0 + half * 16];
    v16bf a0 = frag_a_from(ap0);
    v16bf a1 = frag_a_from(ap1);
    v16bf b0 = load_frag_b(B, K, n0,      k0, lane);
    v16bf b1 = load_frag_b(B, K, n0 + 16, k0, lane);
    acc00 = wmma_bf16(a0, b0, acc00);
    acc01 = wmma_bf16(a0, b1, acc01);
    acc10 = wmma_bf16(a1, b0, acc10);
    acc11 = wmma_bf16(a1, b1, acc11);
  }

  // C/D layout: lane l holds col n0+(l&15); rows m_tile + v + (l>=16 ? 8 : 0)
  const int rbase = half << 3;
  float* c00 = C + (size_t)(m0 + rbase) * N + n0 + l16;
  float* c10 = C + (size_t)(m0 + 16 + rbase) * N + n0 + l16;
  #pragma unroll
  for (int v = 0; v < 8; ++v) {
    c00[(size_t)v * N]      = acc00[v];
    c00[(size_t)v * N + 16] = acc01[v];
    c10[(size_t)v * N]      = acc10[v];
    c10[(size_t)v * N + 16] = acc11[v];
  }
}

// --- pointwise LSTM gate kernel (shared by both cells) ---------------------
__global__ void lstm_gates(const float* __restrict__ G,
                           const float* __restrict__ b_ih,
                           const float* __restrict__ b_hh,
                           float* __restrict__ c,
                           bf16_t* __restrict__ hdst1, int ld1, int off1,
                           bf16_t* __restrict__ hdst2, int ld2, int off2,
                           float* __restrict__ hf32, int width) {
  const int idx = blockIdx.x * blockDim.x + threadIdx.x;  // grid sized exactly
  const int n = idx / width;
  const int j = idx - n * width;
  const float* g = G + (size_t)n * 4 * width;
  float gi = g[j]             + b_ih[j]             + b_hh[j];
  float gf = g[width + j]     + b_ih[width + j]     + b_hh[width + j];
  float gg = g[2 * width + j] + b_ih[2 * width + j] + b_hh[2 * width + j];
  float go = g[3 * width + j] + b_ih[3 * width + j] + b_hh[3 * width + j];
  float cn = sigf(gf) * c[idx] + sigf(gi) * tanhf(gg);
  float hn = sigf(go) * tanhf(cn);
  c[idx] = cn;
  bf16_t hb = f2bf(hn);
  if (hdst1) hdst1[(size_t)n * ld1 + off1 + j] = hb;
  if (hdst2) hdst2[(size_t)n * ld2 + off2 + j] = hb;
  if (hf32)  hf32[idx] = hn;
}

// --- attention: energy -> masked softmax -> context ------------------------
// keyT: [n][k][t] (transposed for coalesced energy), valB: [n][t][v].
__global__ void attention_kernel(const bf16_t* __restrict__ keyT,
                                 const bf16_t* __restrict__ valB,
                                 const bf16_t* __restrict__ act2,
                                 const int* __restrict__ lens,
                                 float* __restrict__ ctxf,
                                 bf16_t* __restrict__ act1) {
  __shared__ float h2s[KSZ];
  __shared__ float att[TENC];
  __shared__ float red[TENC];
  const int n = blockIdx.x;
  const int tid = threadIdx.x;  // 512 threads
  if (tid < KSZ) h2s[tid] = bf2f(act2[(size_t)n * K2 + HSZ + tid]);
  __syncthreads();

  const bf16_t* kp = keyT + (size_t)n * KSZ * TENC + tid;
  float e = 0.0f;
  #pragma unroll 4
  for (int k = 0; k < KSZ; ++k) e += h2s[k] * bf2f(kp[(size_t)k * TENC]);
  const int lens8 = lens[n] >> 3;
  if (tid >= lens8) e = -1.0e9f;
  att[tid] = e; red[tid] = e;
  __syncthreads();
  for (int s = TENC / 2; s > 0; s >>= 1) {
    if (tid < s) red[tid] = fmaxf(red[tid], red[tid + s]);
    __syncthreads();
  }
  const float m = red[0];
  __syncthreads();
  float ex = __expf(att[tid] - m);
  att[tid] = ex; red[tid] = ex;
  __syncthreads();
  for (int s = TENC / 2; s > 0; s >>= 1) {
    if (tid < s) red[tid] += red[tid + s];
    __syncthreads();
  }
  const float inv = 1.0f / red[0];
  __syncthreads();
  att[tid] *= inv;
  __syncthreads();

  if (tid < VSZ) {
    const bf16_t* vp = valB + (size_t)n * TENC * VSZ + tid;
    float acc = 0.0f;
    #pragma unroll 4
    for (int t = 0; t < TENC; ++t) acc += att[t] * bf2f(vp[(size_t)t * VSZ]);
    ctxf[(size_t)n * VSZ + tid] = acc;
    act1[(size_t)n * K1 + HSZ + tid] = f2bf(acc);  // ctx slot for next step
  }
}

// --- tiny output projection (35 logits) ------------------------------------
__global__ void out_proj(const float* __restrict__ h2f,
                         const float* __restrict__ ctxf,
                         const float* __restrict__ Wout,
                         const float* __restrict__ bout,
                         float* __restrict__ out, int step) {
  const int idx = blockIdx.x * blockDim.x + threadIdx.x;
  if (idx >= NB * VOC) return;
  const int n = idx / VOC;
  const int v = idx - n * VOC;
  const float* w  = Wout + (size_t)v * (KSZ + VSZ);
  const float* h  = h2f  + (size_t)n * KSZ;
  const float* cx = ctxf + (size_t)n * VSZ;
  float s = bout[v];
  #pragma unroll 4
  for (int j = 0; j < KSZ; ++j) s += h[j] * w[j];
  #pragma unroll 4
  for (int j = 0; j < VSZ; ++j) s += cx[j] * w[KSZ + j];
  out[(size_t)n * STEPS * VOC + (size_t)step * VOC + v] = s;
}

// --- setup / conversion kernels --------------------------------------------
__global__ void prep_x(const float* __restrict__ emb, const int* __restrict__ text,
                       bf16_t* __restrict__ act1, int step) {
  const int idx = blockIdx.x * blockDim.x + threadIdx.x;  // NB*HSZ
  const int n = idx >> 9;
  const int j = idx & 511;
  const int tok = text[n * TEXTL + step];
  act1[(size_t)n * K1 + j] = f2bf(emb[(size_t)tok * HSZ + j]);
}

__global__ void conv_wcat1(const float* __restrict__ Wih, const float* __restrict__ Whh,
                           bf16_t* __restrict__ out) {
  const int idx = blockIdx.x * blockDim.x + threadIdx.x;  // N1*K1
  const int row = idx / K1;
  const int k   = idx - row * K1;
  float v = (k < HSZ + VSZ) ? Wih[(size_t)row * (HSZ + VSZ) + k]
                            : Whh[(size_t)row * HSZ + (k - (HSZ + VSZ))];
  out[idx] = f2bf(v);
}

__global__ void conv_wcat2(const float* __restrict__ Wih, const float* __restrict__ Whh,
                           bf16_t* __restrict__ out) {
  const int idx = blockIdx.x * blockDim.x + threadIdx.x;  // N2*K2
  const int row = idx / K2;
  const int k   = idx - row * K2;
  float v = (k < HSZ) ? Wih[(size_t)row * HSZ + k]
                      : Whh[(size_t)row * KSZ + (k - HSZ)];
  out[idx] = f2bf(v);
}

__global__ void conv_keys(const float* __restrict__ key, bf16_t* __restrict__ keyT) {
  const size_t idx = (size_t)blockIdx.x * blockDim.x + threadIdx.x;  // NB*KSZ*TENC
  const int t = (int)(idx & (TENC - 1));
  const int k = (int)((idx >> 9) & (KSZ - 1));
  const int n = (int)(idx >> 17);
  keyT[idx] = f2bf(key[(size_t)n * TENC * KSZ + (size_t)t * KSZ + k]);
}

__global__ void conv_vals(const float* __restrict__ val, bf16_t* __restrict__ out) {
  const size_t idx = (size_t)blockIdx.x * blockDim.x + threadIdx.x;
  out[idx] = f2bf(val[idx]);
}

__global__ void zero_ws(unsigned* __restrict__ p, int words) {
  const int idx = blockIdx.x * blockDim.x + threadIdx.x;
  if (idx < words) p[idx] = 0u;
}

// --- workspace layout (bytes) ----------------------------------------------
#define OFF_WCAT1 0
#define SZ_WCAT1  ((size_t)N1 * K1 * 2)                    // 5,242,880
#define OFF_WCAT2 (OFF_WCAT1 + SZ_WCAT1)
#define SZ_WCAT2  ((size_t)N2 * K2 * 2)                    // 1,572,864
#define OFF_KEYT  (OFF_WCAT2 + SZ_WCAT2)
#define SZ_KEYT   ((size_t)NB * KSZ * TENC * 2)            // 67,108,864
#define OFF_VALB  (OFF_KEYT + SZ_KEYT)
#define SZ_VALB   ((size_t)NB * TENC * VSZ * 2)            // 67,108,864
#define OFF_ACT1  (OFF_VALB + SZ_VALB)
#define SZ_ACT1   ((size_t)NB * K1 * 2)                    // 655,360
#define OFF_ACT2  (OFF_ACT1 + SZ_ACT1)
#define SZ_ACT2   ((size_t)NB * K2 * 2)                    // 393,216
#define OFF_C1    (OFF_ACT2 + SZ_ACT2)
#define SZ_C1     ((size_t)NB * HSZ * 4)                   // 524,288
#define OFF_C2    (OFF_C1 + SZ_C1)
#define SZ_C2     ((size_t)NB * KSZ * 4)                   // 262,144
#define OFF_H2F   (OFF_C2 + SZ_C2)
#define SZ_H2F    ((size_t)NB * KSZ * 4)
#define OFF_CTXF  (OFF_H2F + SZ_H2F)
#define SZ_CTXF   ((size_t)NB * VSZ * 4)
#define OFF_G1    (OFF_CTXF + SZ_CTXF)
#define SZ_G1     ((size_t)NB * N1 * 4)                    // 2,097,152
#define OFF_G2    (OFF_G1 + SZ_G1)
#define SZ_G2     ((size_t)NB * N2 * 4)
#define ZERO_WORDS ((int)((SZ_ACT1 + SZ_ACT2 + SZ_C1 + SZ_C2 + SZ_H2F + SZ_CTXF) / 4))

extern "C" void kernel_launch(void* const* d_in, const int* in_sizes, int n_in,
                              void* d_out, int out_size, void* d_ws, size_t ws_size,
                              hipStream_t stream) {
  (void)in_sizes; (void)n_in; (void)out_size; (void)ws_size;
  const float* enc_key = (const float*)d_in[0];
  const float* enc_val = (const float*)d_in[1];
  const int*   text    = (const int*)d_in[2];
  const int*   lens    = (const int*)d_in[3];
  const float* emb     = (const float*)d_in[5];
  const float* W_ih1   = (const float*)d_in[6];
  const float* W_hh1   = (const float*)d_in[7];
  const float* b_ih1   = (const float*)d_in[8];
  const float* b_hh1   = (const float*)d_in[9];
  const float* W_ih2   = (const float*)d_in[10];
  const float* W_hh2   = (const float*)d_in[11];
  const float* b_ih2   = (const float*)d_in[12];
  const float* b_hh2   = (const float*)d_in[13];
  const float* W_out   = (const float*)d_in[14];
  const float* b_out   = (const float*)d_in[15];

  char* ws = (char*)d_ws;
  bf16_t* wcat1 = (bf16_t*)(ws + OFF_WCAT1);
  bf16_t* wcat2 = (bf16_t*)(ws + OFF_WCAT2);
  bf16_t* keyT  = (bf16_t*)(ws + OFF_KEYT);
  bf16_t* valB  = (bf16_t*)(ws + OFF_VALB);
  bf16_t* act1  = (bf16_t*)(ws + OFF_ACT1);
  bf16_t* act2  = (bf16_t*)(ws + OFF_ACT2);
  float*  c1    = (float*)(ws + OFF_C1);
  float*  c2    = (float*)(ws + OFF_C2);
  float*  h2f   = (float*)(ws + OFF_H2F);
  float*  ctxf  = (float*)(ws + OFF_CTXF);
  float*  G1    = (float*)(ws + OFF_G1);
  float*  G2    = (float*)(ws + OFF_G2);
  float*  out   = (float*)d_out;

  // One-time (per call) conversions: weights -> bf16 (concatenated), enc
  // key/value -> bf16 (keys transposed to [n][k][t]); state zeroed.
  conv_wcat1<<<(N1 * K1) / 256, 256, 0, stream>>>(W_ih1, W_hh1, wcat1);
  conv_wcat2<<<(N2 * K2) / 256, 256, 0, stream>>>(W_ih2, W_hh2, wcat2);
  conv_keys <<<(NB * KSZ * TENC) / 256, 256, 0, stream>>>(enc_key, keyT);
  conv_vals <<<(NB * TENC * VSZ) / 256, 256, 0, stream>>>(enc_val, valB);
  zero_ws   <<<(ZERO_WORDS + 255) / 256, 256, 0, stream>>>((unsigned*)(ws + OFF_ACT1), ZERO_WORDS);

  const int gemm1_blocks = (NB / 32) * (N1 / 256);  // 8*8 = 64
  const int gemm2_blocks = (NB / 32) * (N2 / 256);  // 8*4 = 32

  for (int t = 0; t < STEPS; ++t) {
    prep_x<<<(NB * HSZ) / 256, 256, 0, stream>>>(emb, text, act1, t);
    // LSTM1: G1 = [x|ctx|h1] (256x1280) * Wcat1(2048x1280)^T
    wmma_gemm_lds<K1><<<gemm1_blocks, 256, 0, stream>>>(act1, wcat1, G1, N1);
    lstm_gates<<<(NB * HSZ) / 256, 256, 0, stream>>>(G1, b_ih1, b_hh1, c1,
                                                     act1, K1, HSZ + VSZ,
                                                     act2, K2, 0, nullptr, HSZ);
    // LSTM2: G2 = [h1|h2] (256x768) * Wcat2(1024x768)^T
    wmma_gemm_lds<K2><<<gemm2_blocks, 256, 0, stream>>>(act2, wcat2, G2, N2);
    lstm_gates<<<(NB * KSZ) / 256, 256, 0, stream>>>(G2, b_ih2, b_hh2, c2,
                                                     act2, K2, HSZ,
                                                     nullptr, 0, 0, h2f, KSZ);
    attention_kernel<<<NB, TENC, 0, stream>>>(keyT, valB, act2, lens, ctxf, act1);
    out_proj<<<(NB * VOC + 127) / 128, 128, 0, stream>>>(h2f, ctxf, W_out, b_out,
                                                         out, t);
  }
}
